// PSIWithHooks_14053132992741
// MI455X (gfx1250) — compile-verified, matrix operand order checked by reading, not run on hardware
//
#include <hip/hip_runtime.h>
#include <hip/hip_bf16.h>

typedef __attribute__((ext_vector_type(16))) __bf16 v16bf;
typedef __attribute__((ext_vector_type(8)))  float  v8f;

#define DD        1024
#define BB        4
#define SS        2048
#define MROWS     (BB * SS)     /* 8192 */
#define NCAT      (5 * DD)      /* 5120 */
#define CTXN      (4 * DD)      /* 4096 */
#define MAGSCALE  5.0f
#define ASTRIDE   40            /* padded LDS row stride in ushorts */
#define NCHUNK    16
#define CHUNK     (SS / NCHUNK) /* 128 */
#define NCH       (BB * DD)     /* 4096 channels */

__device__ __forceinline__ unsigned short f2bf(float f) {
    unsigned int u = __float_as_uint(f);
    unsigned int r = u + 0x7fffu + ((u >> 16) & 1u);   // round-to-nearest-even
    return (unsigned short)(r >> 16);
}

// ---------------- conversion kernels ----------------

__global__ void k_conv_f32_to_bf16(const float* __restrict__ src,
                                   unsigned short* __restrict__ dst, int n) {
    for (int i = blockIdx.x * blockDim.x + threadIdx.x; i < n;
         i += gridDim.x * blockDim.x)
        dst[i] = f2bf(src[i]);
}

__global__ void k_conv_wcat(const float* __restrict__ w0, const float* __restrict__ w1,
                            const float* __restrict__ w2, const float* __restrict__ w3,
                            const float* __restrict__ w4, unsigned short* __restrict__ wt) {
    const int total = 5 * DD * DD;
    for (int idx = blockIdx.x * blockDim.x + threadIdx.x; idx < total;
         idx += gridDim.x * blockDim.x) {
        int o = idx / DD;          // output column 0..5119
        int i = idx % DD;          // input dim
        int k = o / DD, oo = o % DD;
        const float* w = (k == 0) ? w0 : (k == 1) ? w1 : (k == 2) ? w2 : (k == 3) ? w3 : w4;
        wt[idx] = f2bf(w[(size_t)i * DD + oo]);
    }
}

__global__ void k_conv_wout(const float* __restrict__ w, unsigned short* __restrict__ wt) {
    const int total = DD * CTXN;
    for (int idx = blockIdx.x * blockDim.x + threadIdx.x; idx < total;
         idx += gridDim.x * blockDim.x) {
        int o = idx / CTXN;
        int i = idx % CTXN;
        wt[idx] = f2bf(w[(size_t)i * DD + o]);
    }
}

__global__ void k_conv_bcat(const float* __restrict__ b0, const float* __restrict__ b1,
                            const float* __restrict__ b2, const float* __restrict__ b3,
                            const float* __restrict__ b4, float* __restrict__ bc) {
    for (int idx = blockIdx.x * blockDim.x + threadIdx.x; idx < NCAT;
         idx += gridDim.x * blockDim.x) {
        int k = idx / DD, o = idx % DD;
        const float* b = (k == 0) ? b0 : (k == 1) ? b1 : (k == 2) ? b2 : (k == 3) ? b3 : b4;
        bc[idx] = b[o];
    }
}

// ---------------- WMMA GEMM with async-to-LDS double buffering ----------------
// C[M][N] = A[M][K](bf16) * Bt[N][K](bf16)^T + bias[N]  (+ residual if mode==1)
// 256 threads = 8 waves (4 M x 2 N), block tile 128x128, wave tile 32x64,
// K-step 32, v_wmma_f32_16x16x32_bf16; tiles staged via global_load_async_to_lds_b128.

#define ASYNC_B128(ldsaddr, gaddr) \
    asm volatile("global_load_async_to_lds_b128 %0, %1, off" \
                 :: "v"(ldsaddr), "v"(gaddr) : "memory")

__global__ __launch_bounds__(256)
void k_gemm_wmma(const unsigned short* __restrict__ A,
                 const unsigned short* __restrict__ Bt,
                 const float* __restrict__ bias,
                 float* __restrict__ Cout,
                 const float* __restrict__ Xres,
                 int Ktot, int Ntot, int mode) {
    __shared__ alignas(16) unsigned short Asl[2][128 * ASTRIDE];
    __shared__ alignas(16) unsigned short Bsl[2][128 * ASTRIDE];

    const int tid  = threadIdx.x;
    const int lane = tid & 31;
    const int wave = tid >> 5;
    const int wm = wave & 3;       // M sub-tile (32 rows)
    const int wn = wave >> 2;      // N sub-tile (64 cols)
    const int m0 = blockIdx.y * 128;
    const int n0 = blockIdx.x * 128;

    // staging assignment: each thread owns A-uint4 {tid, tid+256}, B-uint4 {tid, tid+256}
    const int r0 = tid >> 2, r1 = r0 + 64;
    const int cc = tid & 3;
    unsigned long long ga0 = (unsigned long long)(uintptr_t)A  + ((size_t)(m0 + r0) * Ktot + cc * 8) * 2;
    unsigned long long ga1 = (unsigned long long)(uintptr_t)A  + ((size_t)(m0 + r1) * Ktot + cc * 8) * 2;
    unsigned long long gb0 = (unsigned long long)(uintptr_t)Bt + ((size_t)(n0 + r0) * Ktot + cc * 8) * 2;
    unsigned long long gb1 = (unsigned long long)(uintptr_t)Bt + ((size_t)(n0 + r1) * Ktot + cc * 8) * 2;
    const unsigned lA0 = (unsigned)(uintptr_t)&Asl[0][r0 * ASTRIDE + cc * 8];
    const unsigned lA1 = (unsigned)(uintptr_t)&Asl[0][r1 * ASTRIDE + cc * 8];
    const unsigned lB0 = (unsigned)(uintptr_t)&Bsl[0][r0 * ASTRIDE + cc * 8];
    const unsigned lB1 = (unsigned)(uintptr_t)&Bsl[0][r1 * ASTRIDE + cc * 8];
    const unsigned bufstep = 128 * ASTRIDE * 2;   // bytes between double buffers

    v8f acc[2][4];
#pragma unroll
    for (int a = 0; a < 2; ++a)
#pragma unroll
        for (int b = 0; b < 4; ++b) acc[a][b] = (v8f){};

    const int lsub  = lane & 15;
    const int khalf = (lane >> 4) * 8;     // lanes 16-31 start at K=8 (ISA 7.12.2)
    int koff[8];
#pragma unroll
    for (int j = 0; j < 8; ++j)
        koff[j] = khalf + ((j < 4) ? 2 * j : 2 * j + 8);  // K pairs 0..7, then +16

    const int nk = Ktot >> 5;

    // prologue: async-stage tile 0 into buffer 0 (4 ops/thread)
    ASYNC_B128(lA0, ga0); ASYNC_B128(lA1, ga1);
    ASYNC_B128(lB0, gb0); ASYNC_B128(lB1, gb1);
    ga0 += 64; ga1 += 64; gb0 += 64; gb1 += 64;

    for (int i = 0; i < nk; ++i) {
        const int cur = i & 1;
        if (i + 1 < nk) {
            const unsigned boff = (cur ^ 1) * bufstep;
            ASYNC_B128(lA0 + boff, ga0); ASYNC_B128(lA1 + boff, ga1);
            ASYNC_B128(lB0 + boff, gb0); ASYNC_B128(lB1 + boff, gb1);
            ga0 += 64; ga1 += 64; gb0 += 64; gb1 += 64;
            asm volatile("s_wait_asynccnt 0x4" ::: "memory");  // tile i landed, i+1 in flight
        } else {
            asm volatile("s_wait_asynccnt 0x0" ::: "memory");
        }
        __syncthreads();

        union Frag { v16bf v; unsigned int u[8]; };
        Frag af[2], bf[4];
        const unsigned short* Ab = Asl[cur];
        const unsigned short* Bb = Bsl[cur];
#pragma unroll
        for (int t = 0; t < 2; ++t) {
            const int m = wm * 32 + t * 16 + lsub;
#pragma unroll
            for (int j = 0; j < 8; ++j)
                af[t].u[j] = *(const unsigned int*)&Ab[m * ASTRIDE + koff[j]];
        }
#pragma unroll
        for (int t = 0; t < 4; ++t) {
            const int n = wn * 64 + t * 16 + lsub;
#pragma unroll
            for (int j = 0; j < 8; ++j)
                bf[t].u[j] = *(const unsigned int*)&Bb[n * ASTRIDE + koff[j]];
        }
#pragma unroll
        for (int a = 0; a < 2; ++a)
#pragma unroll
            for (int b = 0; b < 4; ++b)
                acc[a][b] = __builtin_amdgcn_wmma_f32_16x16x32_bf16(
                    false, af[a].v, false, bf[b].v, (short)0, acc[a][b], false, false);
        __syncthreads();
    }

    // epilogue: C layout — VGPR r holds M=r (lanes 0-15) / M=r+8 (lanes 16-31), N=lane%16
    const int radd = (lane >> 4) * 8;
#pragma unroll
    for (int b = 0; b < 4; ++b) {
        const int col = n0 + wn * 64 + b * 16 + lsub;
        const float bv = bias[col];
#pragma unroll
        for (int a = 0; a < 2; ++a) {
#pragma unroll
            for (int r = 0; r < 8; ++r) {
                const int row = m0 + wm * 32 + a * 16 + r + radd;
                float v = acc[a][b][r] + bv;
                if (mode == 1) v += Xres[(size_t)row * Ntot + col];
                Cout[(size_t)row * Ntot + col] = v;
            }
        }
    }
}

// ---------------- hierarchical sequence scan ----------------
// S split into NCHUNK chunks; additive scans done as chunk-partials + per-channel
// exclusive scan + chunk walk. phi must be resolved first, then memory scans.

__device__ __forceinline__ void ch_decode(int t, int& b, int& c, int& d) {
    b = t >> 14;            // / (NCHUNK*DD)
    c = (t >> 10) & 15;     // chunk
    d = t & 1023;           // channel dim
}

// A: chunk-local sums of gated_omega
__global__ __launch_bounds__(256)
void k_scanA(const float* __restrict__ Y, const float* __restrict__ iscale,
             float* __restrict__ part_go) {
    int t = blockIdx.x * blockDim.x + threadIdx.x;
    if (t >= BB * NCHUNK * DD) return;
    int b, c, d; ch_decode(t, b, c, d);
    const float isc = fabsf(iscale[d]);
    float sum = 0.f;
    for (int s = c * CHUNK; s < (c + 1) * CHUNK; ++s) {
        const float* yr = Y + ((size_t)b * SS + s) * NCAT;
        const float omega    = yr[d];
        const float gate_pre = yr[3 * DD + d];
        sum += omega * isc / (1.f + __expf(-gate_pre));
    }
    part_go[(b * NCHUNK + c) * DD + d] = sum;
}

// B: per-channel exclusive scan over chunks (in place), 1 array
__global__ __launch_bounds__(256)
void k_scanB(float* __restrict__ p) {
    int ch = blockIdx.x * blockDim.x + threadIdx.x;
    if (ch >= NCH) return;
    int b = ch >> 10, d = ch & 1023;
    float run = 0.f;
    for (int c = 0; c < NCHUNK; ++c) {
        size_t idx = (size_t)(b * NCHUNK + c) * DD + d;
        float v = p[idx];
        p[idx] = run;
        run += v;
    }
}

// C: resolve phi per element (store), chunk-local sums of memory contributions
__global__ __launch_bounds__(256)
void k_scanC(const float* __restrict__ Y, const float* __restrict__ x,
             const float* __restrict__ iscale, const float* __restrict__ off_go,
             float* __restrict__ phiBuf,
             float* __restrict__ part_r, float* __restrict__ part_i,
             float* __restrict__ part_m) {
    int t = blockIdx.x * blockDim.x + threadIdx.x;
    if (t >= BB * NCHUNK * DD) return;
    int b, c, d; ch_decode(t, b, c, d);
    const float isc = fabsf(iscale[d]);
    float phi_run = off_go[(b * NCHUNK + c) * DD + d];
    float sr = 0.f, si = 0.f, sm = 0.f;
    for (int s = c * CHUNK; s < (c + 1) * CHUNK; ++s) {
        const size_t row = (size_t)b * SS + s;
        const float* yr = Y + row * NCAT;
        const float omega    = yr[d];
        const float mag_pre  = yr[DD + d];
        const float phi0     = yr[2 * DD + d];
        const float gate_pre = yr[3 * DD + d];
        phi_run += omega * isc / (1.f + __expf(-gate_pre));
        const float phi = phi0 + phi_run;
        phiBuf[row * DD + d] = phi;
        float sp, cp;
        __sincosf(phi, &sp, &cp);
        const float magnitude = MAGSCALE / (1.f + __expf(-mag_pre));
        const float wc = magnitude * x[row * DD + d];
        sr += wc * cp;
        si += wc * sp;
        sm += magnitude;
    }
    const size_t idx = (size_t)(b * NCHUNK + c) * DD + d;
    part_r[idx] = sr; part_i[idx] = si; part_m[idx] = sm;
}

// D: per-channel exclusive scans of the 3 memory partial arrays
__global__ __launch_bounds__(256)
void k_scanD(float* __restrict__ pr, float* __restrict__ pi, float* __restrict__ pm) {
    int ch = blockIdx.x * blockDim.x + threadIdx.x;
    if (ch >= NCH) return;
    int b = ch >> 10, d = ch & 1023;
    float rr = 0.f, ri = 0.f, rm = 0.f;
    for (int c = 0; c < NCHUNK; ++c) {
        size_t idx = (size_t)(b * NCHUNK + c) * DD + d;
        float vr = pr[idx], vi = pi[idx], vm = pm[idx];
        pr[idx] = rr; pi[idx] = ri; pm[idx] = rm;
        rr += vr; ri += vi; rm += vm;
    }
}

// E: final chunk walk — running memory state, retrieval, bf16 context
__global__ __launch_bounds__(256)
void k_scanE(const float* __restrict__ Y, const float* __restrict__ x,
             const float* __restrict__ phiBuf,
             const float* __restrict__ off_r, const float* __restrict__ off_i,
             const float* __restrict__ off_m,
             unsigned short* __restrict__ ctx) {
    int t = blockIdx.x * blockDim.x + threadIdx.x;
    if (t >= BB * NCHUNK * DD) return;
    int b, c, d; ch_decode(t, b, c, d);
    const size_t pidx = (size_t)(b * NCHUNK + c) * DD + d;
    float mr = off_r[pidx], mi = off_i[pidx], macc = off_m[pidx];
    for (int s = c * CHUNK; s < (c + 1) * CHUNK; ++s) {
        const size_t row = (size_t)b * SS + s;
        const float* yr = Y + row * NCAT;
        const float mag_pre = yr[DD + d];
        const float q_pre   = yr[4 * DD + d];
        const float xv      = x[row * DD + d];
        const float phi     = phiBuf[row * DD + d];

        float sp, cp;
        __sincosf(phi, &sp, &cp);
        const float magnitude = MAGSCALE / (1.f + __expf(-mag_pre));
        const float wc = magnitude * xv;
        mr   += wc * cp;
        mi   += wc * sp;
        macc += magnitude;
        const float inv = rsqrtf(macc + 1e-8f);
        const float mrn = mr * inv, min_ = mi * inv;

        float sq, cq;
        __sincosf(phi + q_pre, &sq, &cq);
        const float rr = mrn * cq + min_ * sq;
        const float ri = min_ * cq - mrn * sq;

        unsigned short* cr = ctx + row * (size_t)CTXN;
        cr[d]          = f2bf(xv * cp);
        cr[DD + d]     = f2bf(xv * sp);
        cr[2 * DD + d] = f2bf(rr);
        cr[3 * DD + d] = f2bf(ri);
    }
}

// ---------------- launcher ----------------
extern "C" void kernel_launch(void* const* d_in, const int* in_sizes, int n_in,
                              void* d_out, int out_size, void* d_ws, size_t ws_size,
                              hipStream_t stream) {
    const float* x       = (const float*)d_in[0];
    const float* W_omega = (const float*)d_in[1];
    const float* b_omega = (const float*)d_in[2];
    const float* W_mag   = (const float*)d_in[3];
    const float* b_mag   = (const float*)d_in[4];
    const float* W_phi0  = (const float*)d_in[5];
    const float* b_phi0  = (const float*)d_in[6];
    const float* W_gate  = (const float*)d_in[7];
    const float* b_gate  = (const float*)d_in[8];
    const float* W_q     = (const float*)d_in[9];
    const float* b_q     = (const float*)d_in[10];
    const float* iscale  = (const float*)d_in[11];
    const float* W_out   = (const float*)d_in[12];
    const float* b_out   = (const float*)d_in[13];
    float* out = (float*)d_out;

    char* ws = (char*)d_ws;
    size_t off = 0;
    auto carve = [&](size_t bytes) -> void* {
        void* p = ws + off;
        off = (off + bytes + 255) & ~(size_t)255;
        return p;
    };
    unsigned short* xb     = (unsigned short*)carve((size_t)MROWS * DD * 2);     // 16 MB
    unsigned short* wcat   = (unsigned short*)carve((size_t)NCAT * DD * 2);      // 10 MB
    unsigned short* wout   = (unsigned short*)carve((size_t)DD * CTXN * 2);      //  8 MB
    float*          bcat   = (float*)carve((size_t)NCAT * 4);
    float*          Y      = (float*)carve((size_t)MROWS * NCAT * 4);            // 160 MB
    unsigned short* ctx    = (unsigned short*)carve((size_t)MROWS * CTXN * 2);   //  64 MB
    float*          phiBuf = (float*)carve((size_t)MROWS * DD * 4);              //  32 MB
    float*          p_go   = (float*)carve((size_t)NCH * NCHUNK * 4);
    float*          p_r    = (float*)carve((size_t)NCH * NCHUNK * 4);
    float*          p_i    = (float*)carve((size_t)NCH * NCHUNK * 4);
    float*          p_m    = (float*)carve((size_t)NCH * NCHUNK * 4);
    (void)ws_size; (void)in_sizes; (void)n_in; (void)out_size;

    // 1) precision conversion + weight transposition
    k_conv_f32_to_bf16<<<4096, 256, 0, stream>>>(x, xb, MROWS * DD);
    k_conv_wcat<<<4096, 256, 0, stream>>>(W_omega, W_mag, W_phi0, W_gate, W_q, wcat);
    k_conv_wout<<<4096, 256, 0, stream>>>(W_out, wout);
    k_conv_bcat<<<20, 256, 0, stream>>>(b_omega, b_mag, b_phi0, b_gate, b_q, bcat);

    // 2) fused 5-projection GEMM: Y = X * Wcat + bcat    (8192 x 5120, K=1024)
    k_gemm_wmma<<<dim3(NCAT / 128, MROWS / 128), 256, 0, stream>>>(
        xb, wcat, bcat, Y, nullptr, DD, NCAT, 0);

    // 3) hierarchical scans + context build
    const int walkT = BB * NCHUNK * DD;                 // 65536
    k_scanA<<<walkT / 256, 256, 0, stream>>>(Y, iscale, p_go);
    k_scanB<<<NCH / 256, 256, 0, stream>>>(p_go);
    k_scanC<<<walkT / 256, 256, 0, stream>>>(Y, x, iscale, p_go, phiBuf, p_r, p_i, p_m);
    k_scanD<<<NCH / 256, 256, 0, stream>>>(p_r, p_i, p_m);
    k_scanE<<<walkT / 256, 256, 0, stream>>>(Y, x, phiBuf, p_r, p_i, p_m, ctx);

    // 4) output GEMM with residual: out = x + ctx * W_out + b_out  (8192 x 1024, K=4096)
    k_gemm_wmma<<<dim3(DD / 128, MROWS / 128), 256, 0, stream>>>(
        ctx, wout, b_out, out, x, CTXN, DD, 1);
}
